// MambaBlock_36627481100334
// MI455X (gfx1250) — compile-verified, matrix operand order checked by reading, not run on hardware
//
#include <hip/hip_runtime.h>
#include <hip/hip_bf16.h>

// ---------------------------------------------------------------------------
// Mamba block for MI455X (gfx1250, wave32, WMMA, 192MB L2, 320KB LDS/WGP).
//
//  * GEMMs: v_wmma_f32_16x16x32_bf16, f32 accumulate. Operands pre-converted
//    to bf16 once per launch; weights pre-transposed to [N x K] (zero-padded)
//    so k-loop staging is an unguarded contiguous 128b copy.
//  * Staging uses GLOBAL_LOAD_ASYNC_TO_LDS_B128 (ASYNCcnt) with double-
//    buffered LDS: copy of tile t+1 overlaps WMMA on tile t, zero VGPRs used.
//    Falls back to synchronous b128 staging if the builtins are unavailable.
//  * bf16 operand arrays are L2-resident (192MB), so cross-tile re-reads hit
//    L2; HBM traffic stays ~compulsory.
//  * Selective scan: 65536 lanes over (B, D_INNER, D_STATE), 1 v_exp per
//    lane-step, 16-state reduce via 4 shuffle butterflies.
// ---------------------------------------------------------------------------

#define B_SZ      2
#define L_SEQ     4096
#define D_MODEL   1024
#define D_INNER   2048
#define D_STATE   16
#define D_CONV    4
#define DT_RANK   64
#define NROWS     (B_SZ * L_SEQ)          // 8192

#if __has_builtin(__builtin_amdgcn_global_load_async_to_lds_b128) && \
    __has_builtin(__builtin_amdgcn_s_wait_asynccnt)
#define USE_ASYNC_LDS 1
#else
#define USE_ASYNC_LDS 0
#endif

typedef __attribute__((ext_vector_type(16))) __bf16 bf16x16;
typedef __attribute__((ext_vector_type(8)))  float  v8f;
typedef __attribute__((ext_vector_type(4)))  int    i32x4;

__device__ inline unsigned short f2bf_bits(float f) {
    unsigned u = __builtin_bit_cast(unsigned, f);
    unsigned r = u + 0x7FFFu + ((u >> 16) & 1u);   // round-to-nearest-even
    return (unsigned short)(r >> 16);
}
__device__ inline __bf16 f2bf(float f) {
    return __builtin_bit_cast(__bf16, f2bf_bits(f));
}

__device__ inline float silu_f(float x)     { return x / (1.0f + __expf(-x)); }
__device__ inline float softplus_f(float x) { return (x > 20.0f) ? x : __logf(1.0f + __expf(x)); }

// Fragment load from LDS per ISA 7.12.2 (16-bit A 16x32): lanes 0-15 hold rows
// 0-15 with K 0-7 (v0-3) and 16-23 (v4-7); lanes 16-31 same rows, K+8.
// B tiles are staged pre-transposed [n][k], so the same loader serves both.
#define LDS_LD 40   // padded row stride in bf16 (80B): conflict-free, 16B-aligned
__device__ inline bf16x16 load_frag(const __bf16* base, int lane) {
    const int half = lane >> 4;
    const __bf16* row = base + (lane & 15) * LDS_LD;
    bf16x16 f;
#pragma unroll
    for (int v = 0; v < 8; ++v) {
        int kb = ((v < 4) ? (2 * v) : (16 + 2 * (v - 4))) + 8 * half;
        f[2 * v]     = row[kb];
        f[2 * v + 1] = row[kb + 1];
    }
    return f;
}

#if USE_ASYNC_LDS
// Typed AS-qualified pointer casts (builtin expects int4 addrspace pointers;
// HIP prints AS(1) as "__device__" in diagnostics).
#define ASG(p) ((__attribute__((address_space(1))) i32x4*)(p))
#define ASL(p) ((__attribute__((address_space(3))) i32x4*)(p))
// Copy 16 contiguous bf16 (32B) global -> LDS, 2 async b128 ops, no VGPR data.
#define ASYNC_CP32(gp, lp)                                                    \
    do {                                                                      \
        __builtin_amdgcn_global_load_async_to_lds_b128(ASG(gp), ASL(lp), 0, 0); \
        __builtin_amdgcn_global_load_async_to_lds_b128(ASG(gp), ASL(lp), 16, 0); \
    } while (0)
#endif

// ---------------------------------------------------------------------------
// Tiled GEMM: C[M,N] = A[M,K] @ B[K,N], bf16 operands, f32 accum/output.
//   A  : [M x K] bf16 row-major
//   Bt : [Npad x K] bf16 row-major (pre-transposed weight, zero-padded rows)
// Block tile 128x128, 256 threads = 8 waves (4 M x 2 N), wave tile 32x64
// -> 8 wmma per wave per 32-wide k-step.  Double-buffered async LDS staging.
// EPI: 0 = plain f32 store, 1 = softplus(acc + bias[col]),
//      2 = f32 store + bf16 mirror store to Cbf.
// ---------------------------------------------------------------------------
template <int EPI>
__global__ __launch_bounds__(256)
void gemm_bf16_wmma(const __bf16* __restrict__ A, const __bf16* __restrict__ Bt,
                    float* __restrict__ C, const float* __restrict__ bias,
                    __bf16* __restrict__ Cbf,
                    int M, int N, int K, int lda, int ldb, int ldc)
{
    __shared__ __bf16 As[2][128][LDS_LD];
    __shared__ __bf16 Bs[2][128][LDS_LD];   // [n][k]

    const int tid  = threadIdx.x;
    const int lane = tid & 31;
    const int wave = tid >> 5;
    const int wm   = wave & 3;           // 0..3 -> 32-row sub-tile
    const int wn   = wave >> 2;          // 0..1 -> 64-col group
    const int mb   = blockIdx.y * 128;
    const int nb   = blockIdx.x * 128;

    // Staging: each thread owns 16 contiguous bf16 (32B) of A and of Bt.
    const int frow = tid >> 1;           // 0..127
    const int fcol = (tid & 1) * 16;     // 0 or 16
    const __bf16* gA = A  + (size_t)(mb + frow) * lda + fcol;
    const __bf16* gB = Bt + (size_t)(nb + frow) * ldb + fcol;
    __bf16* lA0 = &As[0][frow][fcol];
    __bf16* lA1 = &As[1][frow][fcol];
    __bf16* lB0 = &Bs[0][frow][fcol];
    __bf16* lB1 = &Bs[1][frow][fcol];

    v8f zero = {};
    v8f acc[2][4];
#pragma unroll
    for (int i = 0; i < 2; ++i)
#pragma unroll
        for (int j = 0; j < 4; ++j) acc[i][j] = zero;

    const int nk = K / 32;

#if USE_ASYNC_LDS
    ASYNC_CP32(gA, lA0);                 // prologue: tile 0 -> buffer 0
    ASYNC_CP32(gB, lB0);
#endif

    for (int t = 0; t < nk; ++t) {
        const int cur = t & 1;
#if USE_ASYNC_LDS
        // Issue tile t+1 into the other buffer (its last readers finished at
        // the barrier that ended iteration t-1), then retire tile t's copies:
        // async loads complete in order, so waiting ASYNCcnt<=4 leaves exactly
        // the 4 just-issued ops in flight.
        if (t + 1 < nk) {
            __bf16* la = cur ? lA0 : lA1;
            __bf16* lb = cur ? lB0 : lB1;
            ASYNC_CP32(gA + (t + 1) * 32, la);
            ASYNC_CP32(gB + (t + 1) * 32, lb);
            __builtin_amdgcn_s_wait_asynccnt(4);
        } else {
            __builtin_amdgcn_s_wait_asynccnt(0);
        }
#else
        {   // synchronous staging fallback
            const int kt = t * 32;
            uint4 a0 = *(const uint4*)(gA + kt);
            uint4 a1 = *(const uint4*)(gA + kt + 8);
            uint4 b0 = *(const uint4*)(gB + kt);
            uint4 b1 = *(const uint4*)(gB + kt + 8);
            *(uint4*)&As[cur][frow][fcol]     = a0;
            *(uint4*)&As[cur][frow][fcol + 8] = a1;
            *(uint4*)&Bs[cur][frow][fcol]     = b0;
            *(uint4*)&Bs[cur][frow][fcol + 8] = b1;
            if (kt + 32 < K) {
                __builtin_prefetch(gA + kt + 32, 0, 1);
                __builtin_prefetch(gB + kt + 32, 0, 1);
            }
        }
#endif
        __syncthreads();

        bf16x16 fa0 = load_frag(&As[cur][wm * 32][0],      lane);
        bf16x16 fa1 = load_frag(&As[cur][wm * 32 + 16][0], lane);
#pragma unroll
        for (int j = 0; j < 4; ++j) {
            bf16x16 fb = load_frag(&Bs[cur][wn * 64 + 16 * j][0], lane);
            acc[0][j] = __builtin_amdgcn_wmma_f32_16x16x32_bf16(
                            false, fa0, false, fb, (short)0, acc[0][j], false, false);
            acc[1][j] = __builtin_amdgcn_wmma_f32_16x16x32_bf16(
                            false, fa1, false, fb, (short)0, acc[1][j], false, false);
        }
        __syncthreads();   // all waves done reading buffer `cur`
    }

    // Epilogue.  D layout: VGPR r holds M = r + 8*(lane>>4), N = lane&15.
    const int half = lane >> 4;
    const int l16  = lane & 15;
#pragma unroll
    for (int i = 0; i < 2; ++i) {
        const int rowb = mb + wm * 32 + i * 16 + half * 8;
#pragma unroll
        for (int j = 0; j < 4; ++j) {
            const int col = nb + wn * 64 + j * 16 + l16;
            if (col < N) {
                const float bcol = (EPI == 1) ? bias[col] : 0.0f;
#pragma unroll
                for (int r = 0; r < 8; ++r) {
                    float v = acc[i][j][r];
                    if (EPI == 1) v = softplus_f(v + bcol);
                    size_t off = (size_t)(rowb + r) * ldc + col;
                    C[off] = v;
                    if (EPI == 2) Cbf[off] = f2bf(v);
                }
            }
        }
    }
}

// ---------------------------------------------------------------------------
// Weight prep: f32 [K x N] -> bf16 [Npad x K] transposed, zero-padded rows.
// grid = (Npad/32, K/32), block = (32, 8).  K, N are multiples of 32.
// ---------------------------------------------------------------------------
__global__ __launch_bounds__(256)
void transpose_w_bf16(const float* __restrict__ in, __bf16* __restrict__ out,
                      int K, int N)
{
    __shared__ float tile[32][33];
    const int nb = blockIdx.x * 32;
    const int kb = blockIdx.y * 32;
    const int tx = threadIdx.x;          // 0..31
    const int ty = threadIdx.y;          // 0..7
#pragma unroll
    for (int i = 0; i < 4; ++i) {
        int k = kb + ty + i * 8;
        int n = nb + tx;
        tile[ty + i * 8][tx] = (n < N) ? in[(size_t)k * N + n] : 0.0f;
    }
    __syncthreads();
#pragma unroll
    for (int i = 0; i < 4; ++i) {
        int n = nb + ty + i * 8;
        int k = kb + tx;
        out[(size_t)n * K + k] = f2bf(tile[tx][ty + i * 8]);
    }
}

// Elementwise f32 -> bf16 (x activations), 4 elems/thread, packed b64 stores.
__global__ __launch_bounds__(256)
void f32_to_bf16_kernel(const float* __restrict__ in, __bf16* __restrict__ out,
                        int n4)
{
    int i = blockIdx.x * 256 + threadIdx.x;
    if (i < n4) {
        float4 v = ((const float4*)in)[i];
        uint2 p;
        p.x = (unsigned)f2bf_bits(v.x) | ((unsigned)f2bf_bits(v.y) << 16);
        p.y = (unsigned)f2bf_bits(v.z) | ((unsigned)f2bf_bits(v.w) << 16);
        ((uint2*)out)[i] = p;
    }
}

// ---------------------------------------------------------------------------
// Depthwise causal conv(4) + bias + SiLU.  Writes f32 (for scan) and bf16
// (GEMM operand).  Interior rows (l>=3) take an unguarded path.
// ---------------------------------------------------------------------------
__global__ __launch_bounds__(256)
void conv_silu_kernel(const float* __restrict__ xz,
                      const float* __restrict__ conv_w,
                      const float* __restrict__ conv_b,
                      float* __restrict__ xc_f32,
                      __bf16* __restrict__ xc_bf)
{
    size_t idx = (size_t)blockIdx.x * 256 + threadIdx.x;   // < 8192*2048
    int    d   = (int)(idx & (D_INNER - 1));
    size_t row = idx >> 11;                                // b*L + l
    int    l   = (int)(row & (L_SEQ - 1));

    float w0 = conv_w[d * D_CONV + 0];
    float w1 = conv_w[d * D_CONV + 1];
    float w2 = conv_w[d * D_CONV + 2];
    float w3 = conv_w[d * D_CONV + 3];

    float acc = conv_b[d];
    const float* base = xz + row * (2 * D_INNER) + d;
    if (l >= 3) {
        acc += base[-3 * 2 * D_INNER] * w0;
        acc += base[-2 * 2 * D_INNER] * w1;
        acc += base[-1 * 2 * D_INNER] * w2;
        acc += base[0] * w3;
    } else {
        if (l >= 2) acc += base[-2 * 2 * D_INNER] * w1;
        if (l >= 1) acc += base[-1 * 2 * D_INNER] * w2;
        acc += base[0] * w3;
    }
    float s = silu_f(acc);
    xc_f32[idx] = s;
    xc_bf[idx]  = f2bf(s);
}

// ---------------------------------------------------------------------------
// Selective scan.  Lane = (channel, state): wave = 2 channels x 16 states.
// B/C loads index by state (same 16 floats for both channels); xc/dt coalesce across d.
// Output y = (scan + xc*D_skip) * silu(z) written directly as bf16 (only the
// final GEMM consumes it).
// ---------------------------------------------------------------------------
__global__ __launch_bounds__(256)
void scan_kernel(const float* __restrict__ xz,
                 const float* __restrict__ dbl,
                 const float* __restrict__ dt,
                 const float* __restrict__ A_log,
                 const float* __restrict__ D_skip,
                 const float* __restrict__ xc_f32,
                 __bf16* __restrict__ y_bf)
{
    int t  = blockIdx.x * 256 + threadIdx.x;  // 0..65535
    int ch = t >> 4;                          // b*D_INNER + d
    int s  = t & 15;
    int b  = ch >> 11;
    int d  = ch & (D_INNER - 1);

    float Acoef = -__expf(A_log[d * D_STATE + s]);
    float dsk   = D_skip[d];
    float h     = 0.0f;
    size_t rowb = (size_t)b * L_SEQ;

    for (int l = 0; l < L_SEQ; ++l) {
        size_t row = rowb + (size_t)l;
        float xcv = xc_f32[row * D_INNER + d];
        float dtv = dt    [row * D_INNER + d];
        float Bv  = dbl   [row * 96 + DT_RANK + s];
        float Cv  = dbl   [row * 96 + DT_RANK + D_STATE + s];

        float dA = __expf(dtv * Acoef);
        h = dA * h + (dtv * xcv) * Bv;
        float y = h * Cv;
        y += __shfl_xor(y, 8, 32);
        y += __shfl_xor(y, 4, 32);
        y += __shfl_xor(y, 2, 32);
        y += __shfl_xor(y, 1, 32);

        if (s == 0) {
            float zv = xz[row * (2 * D_INNER) + D_INNER + d];
            y_bf[row * D_INNER + d] = f2bf((y + xcv * dsk) * silu_f(zv));
        }
    }
}

// ---------------------------------------------------------------------------
// LayerNorm over D_MODEL=1024, one block per row, in place on d_out.
// ---------------------------------------------------------------------------
__global__ __launch_bounds__(256)
void layernorm_kernel(float* __restrict__ out,
                      const float* __restrict__ g,
                      const float* __restrict__ bta)
{
    __shared__ float s1[256], s2[256];
    float* p = out + (size_t)blockIdx.x * D_MODEL;
    int tid = threadIdx.x;

    float v[4], sum = 0.0f, sq = 0.0f;
#pragma unroll
    for (int i = 0; i < 4; ++i) {
        v[i] = p[tid + i * 256];
        sum += v[i];
        sq  += v[i] * v[i];
    }
    s1[tid] = sum; s2[tid] = sq;
    __syncthreads();
    for (int off = 128; off > 0; off >>= 1) {
        if (tid < off) { s1[tid] += s1[tid + off]; s2[tid] += s2[tid + off]; }
        __syncthreads();
    }
    float mu  = s1[0] * (1.0f / D_MODEL);
    float var = s2[0] * (1.0f / D_MODEL) - mu * mu;
    float inv = rsqrtf(var + 1e-5f);
#pragma unroll
    for (int i = 0; i < 4; ++i) {
        int c = tid + i * 256;
        p[c] = (v[i] - mu) * inv * g[c] + bta[c];
    }
}

// ---------------------------------------------------------------------------
extern "C" void kernel_launch(void* const* d_in, const int* in_sizes, int n_in,
                              void* d_out, int out_size, void* d_ws, size_t ws_size,
                              hipStream_t stream)
{
    const float* x      = (const float*)d_in[0];
    const float* W_in   = (const float*)d_in[1];
    const float* conv_w = (const float*)d_in[2];
    const float* conv_b = (const float*)d_in[3];
    const float* W_x    = (const float*)d_in[4];
    const float* W_dt   = (const float*)d_in[5];
    const float* b_dt   = (const float*)d_in[6];
    const float* A_log  = (const float*)d_in[7];
    const float* D_skip = (const float*)d_in[8];
    const float* W_out  = (const float*)d_in[9];
    const float* ln_g   = (const float*)d_in[10];
    const float* ln_b   = (const float*)d_in[11];
    float* out = (float*)d_out;

    // ---- workspace layout (~370 MB) ----
    float* xz   = (float*)d_ws;                              // 8192*4096
    float* xcf  = xz  + (size_t)NROWS * 2 * D_INNER;         // 8192*2048
    float* dtf  = xcf + (size_t)NROWS * D_INNER;             // 8192*2048
    float* dblf = dtf + (size_t)NROWS * D_INNER;             // 8192*96
    __bf16* x_bf   = (__bf16*)(dblf + (size_t)NROWS * 96);   // 8192*1024
    __bf16* xc_bf  = x_bf   + (size_t)NROWS * D_MODEL;       // 8192*2048
    __bf16* y_bf   = xc_bf  + (size_t)NROWS * D_INNER;       // 8192*2048
    __bf16* dbl_bf = y_bf   + (size_t)NROWS * D_INNER;       // 8192*96
    __bf16* Wt_in  = dbl_bf + (size_t)NROWS * 96;            // 4096*1024
    __bf16* Wt_x   = Wt_in  + (size_t)(2 * D_INNER) * D_MODEL; // 128*2048 (padded 96->128)
    __bf16* Wt_dt  = Wt_x   + (size_t)128 * D_INNER;         // 2048*64
    __bf16* Wt_out = Wt_dt  + (size_t)D_INNER * DT_RANK;     // 1024*2048

    dim3 tb(32, 8);
    // ---- one-time (per launch) operand prep: bf16 + transposed weights ----
    f32_to_bf16_kernel<<<(NROWS * D_MODEL / 4 + 255) / 256, 256, 0, stream>>>(
        x, x_bf, NROWS * D_MODEL / 4);
    transpose_w_bf16<<<dim3(4096 / 32, 1024 / 32), tb, 0, stream>>>(W_in,  Wt_in,  D_MODEL, 2 * D_INNER);
    transpose_w_bf16<<<dim3(128 / 32, 2048 / 32),  tb, 0, stream>>>(W_x,   Wt_x,   D_INNER, 96);
    transpose_w_bf16<<<dim3(2048 / 32, 64 / 32),   tb, 0, stream>>>(W_dt,  Wt_dt,  DT_RANK, D_INNER);
    transpose_w_bf16<<<dim3(1024 / 32, 2048 / 32), tb, 0, stream>>>(W_out, Wt_out, D_INNER, D_MODEL);

    // 1) xz = x @ W_in                      (8192x1024)@(1024x4096)
    gemm_bf16_wmma<0><<<dim3(4096 / 128, NROWS / 128), 256, 0, stream>>>(
        x_bf, Wt_in, xz, nullptr, nullptr,
        NROWS, 2 * D_INNER, D_MODEL, D_MODEL, D_MODEL, 2 * D_INNER);

    // 2) causal depthwise conv + SiLU -> xc (f32 + bf16)
    conv_silu_kernel<<<(NROWS * D_INNER) / 256, 256, 0, stream>>>(
        xz, conv_w, conv_b, xcf, xc_bf);

    // 3) dbl = xc @ W_x  (f32 for scan, bf16 mirror for the dt GEMM)
    gemm_bf16_wmma<2><<<dim3(1, NROWS / 128), 256, 0, stream>>>(
        xc_bf, Wt_x, dblf, nullptr, dbl_bf,
        NROWS, 96, D_INNER, D_INNER, D_INNER, 96);

    // 4) dt = softplus(dbl[:, :64] @ W_dt + b_dt)
    gemm_bf16_wmma<1><<<dim3(D_INNER / 128, NROWS / 128), 256, 0, stream>>>(
        dbl_bf, Wt_dt, dtf, b_dt, nullptr,
        NROWS, D_INNER, DT_RANK, 96, DT_RANK, D_INNER);

    // 5) selective scan + D-skip + SiLU(z) gate -> y (bf16)
    scan_kernel<<<(B_SZ * D_INNER * D_STATE) / 256, 256, 0, stream>>>(
        xz, dblf, dtf, A_log, D_skip, xcf, y_bf);

    // 6) out = y @ W_out                    (8192x2048)@(2048x1024)
    gemm_bf16_wmma<0><<<dim3(D_MODEL / 128, NROWS / 128), 256, 0, stream>>>(
        y_bf, Wt_out, out, nullptr, nullptr,
        NROWS, D_MODEL, D_INNER, D_INNER, D_INNER, D_MODEL);

    // 7) LayerNorm in place on out
    layernorm_kernel<<<NROWS, 256, 0, stream>>>(out, ln_g, ln_b);
}
